// BidirectionalMambaBlock_73280732005069
// MI455X (gfx1250) — matrix operand order repaired and checked by reading, not compiled, not run on hardware
//
#include <hip/hip_runtime.h>
#include <hip/hip_bf16.h>
#include <math.h>

// ---------------------------------------------------------------------------
// Types for WMMA fragments (CDNA5 / gfx1250, wave32)
// ---------------------------------------------------------------------------
typedef __attribute__((ext_vector_type(16))) __bf16 v16bf;
typedef __attribute__((ext_vector_type(8)))  __bf16 v8bf;
typedef __attribute__((ext_vector_type(8)))  float  v8f;

// Problem constants
#define BATCH   4
#define SEQ     2048
#define DIM     256
#define DINNER  512
#define DSTATE  16
#define DTRANK  16
#define DCONV   4
#define ROWS    (BATCH * SEQ)          // 8192 "M" rows

__device__ inline v16bf ld_frag2x8(const __bf16* p0, const __bf16* p1) {
    v8bf lo = *(const v8bf*)p0;
    v8bf hi = *(const v8bf*)p1;
    return __builtin_shufflevector(lo, hi, 0,1,2,3,4,5,6,7,8,9,10,11,12,13,14,15);
}

__device__ inline float silu_f(float x) { return x / (1.0f + __expf(-x)); }

// Async global->LDS copy of one 16-byte chunk per lane (ASYNCcnt-tracked).
__device__ inline void async_copy_b128(unsigned lds_addr, const void* gptr) {
    unsigned long long ga = (unsigned long long)(uintptr_t)gptr;
    asm volatile("global_load_async_to_lds_b128 %0, %1, off"
                 :: "v"(lds_addr), "v"(ga) : "memory");
}
__device__ inline void async_wait0() {
    asm volatile("s_wait_asynccnt 0x0" ::: "memory");
}

// ---------------------------------------------------------------------------
// Utility kernels
// ---------------------------------------------------------------------------
__global__ void k_cast_bf16(const float* __restrict__ src, __bf16* __restrict__ dst, int n) {
    int i = blockIdx.x * blockDim.x + threadIdx.x;
    if (i < n) dst[i] = (__bf16)src[i];
}

__global__ void k_copy_f32(const float* __restrict__ src, float* __restrict__ dst, int n) {
    int i = blockIdx.x * blockDim.x + threadIdx.x;
    if (i < n) dst[i] = src[i];
}

// ---------------------------------------------------------------------------
// In-projection: xz[m][n] = sum_k x[b, lsrc, k] * in_w[n][k]
//   M = 8192, N = 1024, K = 256.
//   Block = 8 M-tiles x one 64-col N-group; the 32KB weight block is staged
//   in LDS with GLOBAL_LOAD_ASYNC_TO_LDS_B128 and shared by all 8 waves.
//   Each wave computes a 16x64 strip (4 accumulators) reusing its A fragment.
// ---------------------------------------------------------------------------
__global__ void k_inproj_wmma(const __bf16* __restrict__ xb,
                              const __bf16* __restrict__ wb,
                              float* __restrict__ xz, int rev) {
    __shared__ __bf16 wsm[64 * DIM];             // 32 KB weight block
    const int tid  = threadIdx.x;
    const int wave = tid >> 5;
    const int lane = tid & 31;
    const int g   = blockIdx.x;                  // 64 M-groups * 16 N-groups
    const int mtg = g >> 4;
    const int ng  = g & 15;
    const int mt  = mtg * 8 + wave;
    const int hi  = lane >> 4;
    const int l15 = lane & 15;

    // Stage weight block (contiguous 32 KB): 2048 x b128, 8 per thread.
    {
        const char* src = (const char*)(wb + (size_t)(ng * 64) * DIM);
        unsigned ldsbase = (unsigned)(uintptr_t)(&wsm[0]);
#pragma unroll
        for (int it = 0; it < 8; ++it) {
            int byte_off = (tid + it * 256) * 16;
            async_copy_b128(ldsbase + (unsigned)byte_off, src + byte_off);
        }
        async_wait0();
    }
    __syncthreads();

    // A row for this lane (time-flip applied here for reverse direction)
    const int m = mt * 16 + l15;
    const int b = m >> 11, l = m & (SEQ - 1);
    const int lsrc = rev ? (SEQ - 1 - l) : l;
    const __bf16* arow = xb + (size_t)(b * SEQ + lsrc) * DIM;
    const int kbA = hi * 8;    // A: lanes 0-15 -> K {0..7,16..23}; 16-31 -> {8..15,24..31}
    const int kbB = hi * 8;    // B: lanes 0-15 -> K {0..7,16..23}; 16-31 -> {8..15,24..31}

    const __bf16* brow[4];
#pragma unroll
    for (int j = 0; j < 4; ++j)
        brow[j] = &wsm[(j * 16 + l15) * DIM];    // block-local weight rows (LDS)

    v8f c[4] = {};
#pragma unroll
    for (int k0 = 0; k0 < DIM; k0 += 32) {
        v16bf a = ld_frag2x8(arow + k0 + kbA, arow + k0 + kbA + 16);
#pragma unroll
        for (int j = 0; j < 4; ++j) {
            v16bf bf = ld_frag2x8(brow[j] + k0 + kbB, brow[j] + k0 + kbB + 16);
            c[j] = __builtin_amdgcn_wmma_f32_16x16x32_bf16(false, a, false, bf,
                                                           (short)0, c[j], false, false);
        }
    }
#pragma unroll
    for (int j = 0; j < 4; ++j) {
        const int n = ng * 64 + j * 16 + l15;
#pragma unroll
        for (int r = 0; r < 8; ++r) {
            int mm = mt * 16 + r + hi * 8;
            xz[(size_t)mm * (2 * DINNER) + n] = c[j][r];
        }
    }
}

// ---------------------------------------------------------------------------
// Depthwise causal conv (taps=4) + SiLU.  xc = xz[:, 0:512].
// ---------------------------------------------------------------------------
__global__ void k_conv_silu(const float* __restrict__ xz,
                            const float* __restrict__ cw,
                            const float* __restrict__ cb,
                            float* __restrict__ u) {
    int idx = blockIdx.x * blockDim.x + threadIdx.x;      // ROWS*DINNER
    int d  = idx & (DINNER - 1);
    int bl = idx >> 9;
    int l  = bl & (SEQ - 1);
    float acc = cb[d];
#pragma unroll
    for (int j = 0; j < DCONV; ++j) {
        int lj = l - (DCONV - 1) + j;
        if (lj >= 0)
            acc += cw[d * DCONV + j] * xz[(size_t)(bl - (DCONV - 1) + j) * (2 * DINNER) + d];
    }
    u[idx] = silu_f(acc);
}

// ---------------------------------------------------------------------------
// Fused x-projection (512 -> 48) and dt-projection+softplus (16 -> 512).
// One 64-thread block per row; u row staged in LDS.
// ---------------------------------------------------------------------------
__global__ void k_xproj_dt(const float* __restrict__ u,
                           const float* __restrict__ xpw,   // [48,512]
                           const float* __restrict__ dtw,   // [512,16]
                           const float* __restrict__ dtb,   // [512]
                           float* __restrict__ xdbl,        // [ROWS,48]
                           float* __restrict__ dt) {        // [ROWS,512]
    __shared__ float su[DINNER];
    __shared__ float sx[48];
    const int row = blockIdx.x;
    const int tid = threadIdx.x;                 // 64 threads
    const float* urow = u + (size_t)row * DINNER;
    for (int i = tid; i < DINNER; i += 64) su[i] = urow[i];
    __syncthreads();
    if (tid < 48) {
        float acc = 0.f;
        const float* wr = xpw + tid * DINNER;
#pragma unroll 8
        for (int k = 0; k < DINNER; ++k) acc += su[k] * wr[k];
        sx[tid] = acc;
        xdbl[(size_t)row * 48 + tid] = acc;
    }
    __syncthreads();
#pragma unroll
    for (int i = 0; i < 8; ++i) {
        int dcol = tid * 8 + i;
        float acc = dtb[dcol];
#pragma unroll
        for (int r = 0; r < DTRANK; ++r) acc += sx[r] * dtw[dcol * DTRANK + r];
        dt[(size_t)row * DINNER + dcol] = (acc > 20.f) ? acc : log1pf(__expf(acc));
    }
}

// ---------------------------------------------------------------------------
// Selective scan fused with D-skip and SiLU(z) gating.
// One lane per (b, d, s); 16 states of a channel live in 16 lanes of a wave,
// y reduced with xor-shuffles (bits 0..3 never leave the 16-group).
// Writes gated output directly as bf16 (the out-proj A operand).
// ---------------------------------------------------------------------------
__global__ void k_scan(const float* __restrict__ dt,
                       const float* __restrict__ u,
                       const float* __restrict__ xdbl,
                       const float* __restrict__ xz,        // z = cols 512..1023
                       const float* __restrict__ Alog,
                       const float* __restrict__ Dp,
                       __bf16* __restrict__ yb) {
    const int t = blockIdx.x * blockDim.x + threadIdx.x;    // 32768 lanes
    const int s  = t & (DSTATE - 1);
    const int ch = t >> 4;
    const int d  = ch & (DINNER - 1);
    const int b  = ch >> 9;
    const float A  = -__expf(Alog[d * DSTATE + s]);
    const float Dv = Dp[d];
    float h = 0.f;
    const size_t bl0 = (size_t)b * SEQ;
    for (int l = 0; l < SEQ; ++l) {
        const size_t r = bl0 + l;
        const float dtv = dt[r * DINNER + d];
        const float uv  = u [r * DINNER + d];
        const float Bt  = xdbl[r * 48 + DTRANK + s];
        const float Ct  = xdbl[r * 48 + DTRANK + DSTATE + s];
        const float dA  = __expf(dtv * A);
        h = dA * h + (dtv * uv) * Bt;
        float y = h * Ct;
        y += __shfl_xor(y, 1);
        y += __shfl_xor(y, 2);
        y += __shfl_xor(y, 4);
        y += __shfl_xor(y, 8);
        if (s == 0) {
            const float z = xz[r * (2 * DINNER) + DINNER + d];
            yb[r * DINNER + d] = (__bf16)((y + uv * Dv) * silu_f(z));
        }
    }
}

// ---------------------------------------------------------------------------
// Out-projection with accumulation into d_out (pre-initialized with x):
//   out[b, ldst, n] += sum_k y[m][k] * out_w[n][k]
//   M = 8192, N = 256, K = 512.
//   Block = 8 M-tiles x one 64-col N-group; 64KB weight block async-staged
//   in LDS.  One wave computes a 16x64 strip, C operand = current out tile.
//   rev!=0 writes time-flipped (undo the reverse direction's flip).
// ---------------------------------------------------------------------------
__global__ void k_outproj_wmma(const __bf16* __restrict__ yb,
                               const __bf16* __restrict__ wb,
                               float* __restrict__ out, int rev) {
    __shared__ __bf16 wsm[64 * DINNER];          // 64 KB weight block
    const int tid  = threadIdx.x;
    const int wave = tid >> 5;
    const int lane = tid & 31;
    const int g   = blockIdx.x;                  // 64 M-groups * 4 N-groups
    const int mtg = g >> 2;
    const int ng  = g & 3;
    const int mt  = mtg * 8 + wave;
    const int hi  = lane >> 4;
    const int l15 = lane & 15;

    // Stage weight block (contiguous 64 KB): 4096 x b128, 16 per thread.
    {
        const char* src = (const char*)(wb + (size_t)(ng * 64) * DINNER);
        unsigned ldsbase = (unsigned)(uintptr_t)(&wsm[0]);
#pragma unroll
        for (int it = 0; it < 16; ++it) {
            int byte_off = (tid + it * 256) * 16;
            async_copy_b128(ldsbase + (unsigned)byte_off, src + byte_off);
        }
        async_wait0();
    }
    __syncthreads();

    const int m = mt * 16 + l15;
    const __bf16* arow = yb + (size_t)m * DINNER;
    const int kbA = hi * 8;
    const int kbB = hi * 8;

    const __bf16* brow[4];
#pragma unroll
    for (int j = 0; j < 4; ++j)
        brow[j] = &wsm[(j * 16 + l15) * DINNER]; // block-local weight rows (LDS)

    // Per-lane output row addresses (time-flip applied on write side)
    size_t orow[8];
#pragma unroll
    for (int r = 0; r < 8; ++r) {
        int mm = mt * 16 + r + hi * 8;
        int b = mm >> 11, l = mm & (SEQ - 1);
        int ldst = rev ? (SEQ - 1 - l) : l;
        orow[r] = (size_t)(b * SEQ + ldst) * DIM + l15;
    }

    // Load current out tiles as accumulators (adds x + earlier direction).
    v8f c[4];
#pragma unroll
    for (int j = 0; j < 4; ++j)
#pragma unroll
        for (int r = 0; r < 8; ++r)
            c[j][r] = out[orow[r] + ng * 64 + j * 16];

#pragma unroll
    for (int k0 = 0; k0 < DINNER; k0 += 32) {
        v16bf a = ld_frag2x8(arow + k0 + kbA, arow + k0 + kbA + 16);
#pragma unroll
        for (int j = 0; j < 4; ++j) {
            v16bf bf = ld_frag2x8(brow[j] + k0 + kbB, brow[j] + k0 + kbB + 16);
            c[j] = __builtin_amdgcn_wmma_f32_16x16x32_bf16(false, a, false, bf,
                                                           (short)0, c[j], false, false);
        }
    }
#pragma unroll
    for (int j = 0; j < 4; ++j)
#pragma unroll
        for (int r = 0; r < 8; ++r)
            out[orow[r] + ng * 64 + j * 16] = c[j][r];
}

// ---------------------------------------------------------------------------
// Host launcher
// ---------------------------------------------------------------------------
extern "C" void kernel_launch(void* const* d_in, const int* in_sizes, int n_in,
                              void* d_out, int out_size, void* d_ws, size_t ws_size,
                              hipStream_t stream) {
    (void)in_sizes; (void)n_in; (void)out_size; (void)ws_size;

    const float* x = (const float*)d_in[0];
    float* out = (float*)d_out;

    // Workspace layout (bytes, 256-aligned)
    char* ws = (char*)d_ws;
    const size_t OFF_XBF    = 0;               // 4,194,304  (x as bf16)
    const size_t OFF_INWBF  = 4194304;         //   524,288  (in_w bf16, per dir)
    const size_t OFF_OUTWBF = 4718592;         //   262,144  (out_w bf16, per dir)
    const size_t OFF_XZ     = 4980736;         // 33,554,432 (xz f32 [8192,1024])
    const size_t OFF_U      = 38535168;        // 16,777,216 (u  f32 [8192,512])
    const size_t OFF_XDBL   = 55312384;        //  1,572,864 (x_dbl f32 [8192,48])
    const size_t OFF_DT     = 56885248;        // 16,777,216 (dt f32 [8192,512])
    const size_t OFF_YBF    = 73662464;        //  8,388,608 (gated y bf16 [8192,512])

    __bf16* x_bf   = (__bf16*)(ws + OFF_XBF);
    __bf16* inw_bf = (__bf16*)(ws + OFF_INWBF);
    __bf16* outw_bf= (__bf16*)(ws + OFF_OUTWBF);
    float*  xz     = (float*) (ws + OFF_XZ);
    float*  u      = (float*) (ws + OFF_U);
    float*  xdbl   = (float*) (ws + OFF_XDBL);
    float*  dt     = (float*) (ws + OFF_DT);
    __bf16* y_bf   = (__bf16*)(ws + OFF_YBF);

    const int nX = ROWS * DIM;                 // 2,097,152

    // Shared pre-pass: cast x, init out = x
    k_cast_bf16<<<(nX + 255) / 256, 256, 0, stream>>>(x, x_bf, nX);
    k_copy_f32 <<<(nX + 255) / 256, 256, 0, stream>>>(x, out, nX);

    for (int dir = 0; dir < 2; ++dir) {
        const int base = 1 + dir * 9;
        const float* in_w    = (const float*)d_in[base + 0];   // [1024,256]
        const float* conv_w  = (const float*)d_in[base + 1];   // [512,1,4]
        const float* conv_b  = (const float*)d_in[base + 2];   // [512]
        const float* xproj_w = (const float*)d_in[base + 3];   // [48,512]
        const float* dt_w    = (const float*)d_in[base + 4];   // [512,16]
        const float* dt_b    = (const float*)d_in[base + 5];   // [512]
        const float* A_log   = (const float*)d_in[base + 6];   // [512,16]
        const float* Dp      = (const float*)d_in[base + 7];   // [512]
        const float* out_w   = (const float*)d_in[base + 8];   // [256,512]
        const int rev = dir;                                   // dir 1 = reverse

        k_cast_bf16<<<(2 * DINNER * DIM + 255) / 256, 256, 0, stream>>>(in_w, inw_bf, 2 * DINNER * DIM);
        k_cast_bf16<<<(DIM * DINNER + 255) / 256, 256, 0, stream>>>(out_w, outw_bf, DIM * DINNER);

        // in-proj: 64 M-groups * 16 N-groups = 1024 blocks (8 waves each)
        k_inproj_wmma<<<1024, 256, 0, stream>>>(x_bf, inw_bf, xz, rev);

        // conv + SiLU: 8192*512 threads
        k_conv_silu<<<(ROWS * DINNER) / 256, 256, 0, stream>>>(xz, conv_w, conv_b, u);

        // x-proj + dt-proj: one block per row
        k_xproj_dt<<<ROWS, 64, 0, stream>>>(u, xproj_w, dt_w, dt_b, xdbl, dt);

        // scan + gate: 32768 lanes
        k_scan<<<(BATCH * DINNER * DSTATE) / 256, 256, 0, stream>>>(dt, u, xdbl, xz, A_log, Dp, y_bf);

        // out-proj accumulate: 64 M-groups * 4 N-groups = 256 blocks
        k_outproj_wmma<<<256, 256, 0, stream>>>(y_bf, outw_bf, out, rev);
    }
}
